// M7_59785944760756
// MI455X (gfx1250) — compile-verified, matrix operand order checked by reading, not run on hardware
//
#include <hip/hip_runtime.h>
#include <hip/hip_bf16.h>
#include <math.h>

// ---------------- problem constants ----------------
constexpr int B_  = 2;
constexpr int S_  = 512;
constexpr int E_  = 768;
constexpr int H_  = 8;
constexpr int L_  = 2;
constexpr int V_  = 32000;
constexpr int HE_ = H_ * E_;   // 6144
constexpr int FF_ = 4 * E_;    // 3072
constexpr int MROWS = B_ * S_; // 1024

// ---------------- vector types ----------------
typedef __attribute__((ext_vector_type(16))) _Float16 v16h;
typedef __attribute__((ext_vector_type(8)))  float    v8f;
typedef __attribute__((ext_vector_type(2)))  float    f32x2;
typedef __attribute__((ext_vector_type(4)))  float    f32x4;
typedef __attribute__((ext_vector_type(4)))  int      i32x4;

union Frag { v16h v; f32x4 q[2]; };

// ---------------- CDNA5 async global->LDS path (guarded) ----------------
#if __has_builtin(__builtin_amdgcn_global_load_async_to_lds_b128)
#define HAS_ASYNC 1
#else
#define HAS_ASYNC 0
#endif

#if HAS_ASYNC
__device__ __forceinline__ void async_cp16(const void* g, void* l) {
  // builtin signature (from hipcc diagnostic): (v4i addrspace(1)*, v4i addrspace(3)*, imm, imm)
  __builtin_amdgcn_global_load_async_to_lds_b128(
      (__attribute__((address_space(1))) i32x4*)g,
      (__attribute__((address_space(3))) i32x4*)l, 0, 0);
}
#endif

__device__ __forceinline__ void wait_async0() {
#if __has_builtin(__builtin_amdgcn_s_wait_asynccnt)
  __builtin_amdgcn_s_wait_asynccnt(0);
#else
  asm volatile("s_wait_asynccnt 0x0" ::: "memory");
#endif
}

// ---------------- generic NT WMMA GEMM ----------------
// C[M,N] (op) = scale * A[M,K] * B[N,K]^T ; A,B f16 row-major.
// 256 threads = 8 wave32s. Block tile 64(M) x 128(N), BK = 64.
// Waves laid out 2(M) x 4(N); each wave computes a 32x32 patch = 2x2 WMMA
// tiles -> 8 v_wmma_f32_16x16x32_f16 per K-stage per wave.
// LDS is double buffered; one barrier per K-stage; next stage is brought in
// via GLOBAL_LOAD_ASYNC_TO_LDS (ASYNCcnt) when available, else via registers
// with GLOBAL_PREFETCH of the k+2 tile.
// MODE 0: C f32  = scale*acc
// MODE 1: C f16  = scale*acc * 1/(row+1)   (attention N_reg scaling)
// MODE 2: C f32 += scale*acc               (residual accumulate)
// MODE 3: C f16  = gelu(scale*acc)         (exact erf GELU)
constexpr int BMv = 64, BNv = 128, BKv = 64, LDT = 72;  // 64 K-halves + 8 pad

template <int MODE>
__global__ __launch_bounds__(256) void gemm_nt(
    const _Float16* __restrict__ A, const _Float16* __restrict__ B,
    void* __restrict__ Cv, int M, int N, int K,
    int lda, int ldb, int ldc, float scale)
{
  (void)M; (void)N;
  __shared__ __align__(16) _Float16 As[2][BMv * LDT];
  __shared__ __align__(16) _Float16 Bs[2][BNv * LDT];

  const int t    = threadIdx.x;
  const int lane = t & 31;
  const int r    = lane & 15;
  const int hf   = lane >> 4;
  const int wave = t >> 5;
  const int wM   = wave & 1;   // 0..1 -> 32 rows
  const int wN   = wave >> 1;  // 0..3 -> 32 cols

  const _Float16* Ab = A + (size_t)blockIdx.y * BMv * lda;
  const _Float16* Bb = B + (size_t)blockIdx.x * BNv * ldb;

  // cooperative stage-load indices (8-half = 16B chunks)
  const int ar = t >> 2, ac = (t & 3) * 16;  // A: 64x64, 2 chunks/thread
  const int br = t >> 1, bc = (t & 1) * 32;  // B: 128x64, 4 chunks/thread

  v8f acc[2][2];
#pragma unroll
  for (int i = 0; i < 2; ++i)
#pragma unroll
    for (int j = 0; j < 2; ++j)
      acc[i][j] = (v8f){0.f, 0.f, 0.f, 0.f, 0.f, 0.f, 0.f, 0.f};

  const int NK = K / BKv;

  // ---- prologue: stage 0 into buffer 0 ----
#if HAS_ASYNC
#pragma unroll
  for (int u = 0; u < 2; ++u)
    async_cp16(&Ab[(size_t)ar * lda + ac + 8 * u], &As[0][ar * LDT + ac + 8 * u]);
#pragma unroll
  for (int u = 0; u < 4; ++u)
    async_cp16(&Bb[(size_t)br * ldb + bc + 8 * u], &Bs[0][br * LDT + bc + 8 * u]);
  wait_async0();
#else
  {
    f32x4 a0 = *(const f32x4*)&Ab[(size_t)ar * lda + ac];
    f32x4 a1 = *(const f32x4*)&Ab[(size_t)ar * lda + ac + 8];
    f32x4 b0 = *(const f32x4*)&Bb[(size_t)br * ldb + bc];
    f32x4 b1 = *(const f32x4*)&Bb[(size_t)br * ldb + bc + 8];
    f32x4 b2 = *(const f32x4*)&Bb[(size_t)br * ldb + bc + 16];
    f32x4 b3 = *(const f32x4*)&Bb[(size_t)br * ldb + bc + 24];
    *(f32x4*)&As[0][ar * LDT + ac]      = a0;
    *(f32x4*)&As[0][ar * LDT + ac + 8]  = a1;
    *(f32x4*)&Bs[0][br * LDT + bc]      = b0;
    *(f32x4*)&Bs[0][br * LDT + bc + 8]  = b1;
    *(f32x4*)&Bs[0][br * LDT + bc + 16] = b2;
    *(f32x4*)&Bs[0][br * LDT + bc + 24] = b3;
  }
#endif
  __syncthreads();

  // ---- main loop: one barrier per K-stage, double-buffered LDS ----
  for (int kt = 0; kt < NK; ++kt) {
    const int cur = kt & 1, nxt = cur ^ 1;
    const int kn = (kt + 1) * BKv;
    const bool more = (kt + 1 < NK);

#if HAS_ASYNC
    if (more) {
#pragma unroll
      for (int u = 0; u < 2; ++u)
        async_cp16(&Ab[(size_t)ar * lda + kn + ac + 8 * u],
                   &As[nxt][ar * LDT + ac + 8 * u]);
#pragma unroll
      for (int u = 0; u < 4; ++u)
        async_cp16(&Bb[(size_t)br * ldb + kn + bc + 8 * u],
                   &Bs[nxt][br * LDT + bc + 8 * u]);
    }
#else
    f32x4 a0, a1, b0, b1, b2, b3;
    if (more) {
      a0 = *(const f32x4*)&Ab[(size_t)ar * lda + kn + ac];
      a1 = *(const f32x4*)&Ab[(size_t)ar * lda + kn + ac + 8];
      b0 = *(const f32x4*)&Bb[(size_t)br * ldb + kn + bc];
      b1 = *(const f32x4*)&Bb[(size_t)br * ldb + kn + bc + 8];
      b2 = *(const f32x4*)&Bb[(size_t)br * ldb + kn + bc + 16];
      b3 = *(const f32x4*)&Bb[(size_t)br * ldb + kn + bc + 24];
      if (kt + 2 < NK) {  // CDNA5 global prefetch of the k+2 tile
        __builtin_prefetch(&Ab[(size_t)ar * lda + kn + BKv + ac], 0, 3);
        __builtin_prefetch(&Bb[(size_t)br * ldb + kn + BKv + bc], 0, 3);
      }
    }
#endif

    // compute current stage: 2 K-substeps x (2x2) WMMA with fragment reuse
#pragma unroll
    for (int ks = 0; ks < 2; ++ks) {
      Frag fa[2], fb[2];
#pragma unroll
      for (int i = 0; i < 2; ++i) {
        const _Float16* pa =
            &As[cur][(wM * 32 + 16 * i + r) * LDT + ks * 32 + 8 * hf];
        fa[i].q[0] = *(const f32x4*)pa;
        fa[i].q[1] = *(const f32x4*)(pa + 16);
      }
#pragma unroll
      for (int j = 0; j < 2; ++j) {
        const _Float16* pb =
            &Bs[cur][(wN * 32 + 16 * j + r) * LDT + ks * 32 + 8 * hf];
        fb[j].q[0] = *(const f32x4*)pb;
        fb[j].q[1] = *(const f32x4*)(pb + 16);
      }
#pragma unroll
      for (int i = 0; i < 2; ++i)
#pragma unroll
        for (int j = 0; j < 2; ++j)
          acc[i][j] = __builtin_amdgcn_wmma_f32_16x16x32_f16(
              false, fa[i].v, false, fb[j].v, (short)0, acc[i][j], false, false);
    }

#if HAS_ASYNC
    if (more) wait_async0();
#else
    if (more) {
      *(f32x4*)&As[nxt][ar * LDT + ac]      = a0;
      *(f32x4*)&As[nxt][ar * LDT + ac + 8]  = a1;
      *(f32x4*)&Bs[nxt][br * LDT + bc]      = b0;
      *(f32x4*)&Bs[nxt][br * LDT + bc + 8]  = b1;
      *(f32x4*)&Bs[nxt][br * LDT + bc + 16] = b2;
      *(f32x4*)&Bs[nxt][br * LDT + bc + 24] = b3;
    }
#endif
    __syncthreads();
  }

  // ---- epilogue: C vgpr i -> row (i + 8*hf), col r in each 16x16 tile ----
#pragma unroll
  for (int i = 0; i < 2; ++i) {
#pragma unroll
    for (int j = 0; j < 2; ++j) {
      const int row0 = blockIdx.y * BMv + wM * 32 + i * 16 + hf * 8;
      const int col  = blockIdx.x * BNv + wN * 32 + j * 16 + r;
      if (MODE == 0) {
        float* C = (float*)Cv;
#pragma unroll
        for (int q = 0; q < 8; ++q)
          C[(size_t)(row0 + q) * ldc + col] = acc[i][j][q] * scale;
      } else if (MODE == 1) {
        _Float16* C = (_Float16*)Cv;
#pragma unroll
        for (int q = 0; q < 8; ++q)
          C[(size_t)(row0 + q) * ldc + col] =
              (_Float16)(acc[i][j][q] * scale * (1.0f / (float)(row0 + q + 1)));
      } else if (MODE == 2) {
        float* C = (float*)Cv;
#pragma unroll
        for (int q = 0; q < 8; ++q)
          C[(size_t)(row0 + q) * ldc + col] += acc[i][j][q] * scale;
      } else {
        _Float16* C = (_Float16*)Cv;
#pragma unroll
        for (int q = 0; q < 8; ++q) {
          float v = acc[i][j][q] * scale;
          C[(size_t)(row0 + q) * ldc + col] =
              (_Float16)(0.5f * v * (1.0f + erff(v * 0.70710678118f)));
        }
      }
    }
  }
}

// ---------------- f32 -> f16 convert ----------------
__global__ __launch_bounds__(256) void f2h_kernel(
    const float* __restrict__ in, _Float16* __restrict__ out, long n)
{
  long i = (long)blockIdx.x * blockDim.x + threadIdx.x;
  long stride = (long)gridDim.x * blockDim.x;
  for (; i < n; i += stride) out[i] = (_Float16)in[i];
}

// ---------------- vocab-mean of wte ----------------
__global__ __launch_bounds__(256) void avg_kernel(
    const float* __restrict__ wte, float* __restrict__ avg)
{
  int e = blockIdx.x * 256 + threadIdx.x;
  if (e >= E_) return;
  float s = 0.f;
  for (int v = 0; v < V_; ++v) s += wte[(size_t)v * E_ + e];
  avg[e] = s * (1.0f / (float)V_);
}

// ---------------- embedding ----------------
__global__ __launch_bounds__(256) void embed_kernel(
    const int* __restrict__ x, const float* __restrict__ wte,
    const float* __restrict__ wpe, float* __restrict__ h)
{
  long i = (long)blockIdx.x * 256 + threadIdx.x;  // over B*S*E
  int e  = (int)(i % E_);
  long bs = i / E_;
  int s  = (int)(bs % S_);
  h[i] = wte[(size_t)x[bs] * E_ + e] + wpe[(size_t)s * E_ + e];
}

// ---------------- LayerNorm (bias-free) ----------------
// mode 0: emit per-head Q = xn*w_qk (f16, B,H,S,E) and V^T = xn-avg (f16, B,E,S)
// mode 1: emit plain xn (f16, row-major)
__global__ __launch_bounds__(256) void ln_kernel(
    const float* __restrict__ hbuf, const float* __restrict__ gamma,
    const float* __restrict__ wqk, const float* __restrict__ avg,
    _Float16* __restrict__ outp, _Float16* __restrict__ qh,
    _Float16* __restrict__ vt, int mode)
{
  __shared__ float sd[256];
  const int row = blockIdx.x;        // 0..B*S-1
  const int b = row / S_, s = row % S_;
  const int t = threadIdx.x;
  const float* x = hbuf + (size_t)row * E_;

  float v0 = x[t], v1 = x[t + 256], v2 = x[t + 512];
  sd[t] = v0 + v1 + v2;
  __syncthreads();
  for (int o = 128; o > 0; o >>= 1) { if (t < o) sd[t] += sd[t + o]; __syncthreads(); }
  const float mu = sd[0] * (1.0f / 768.0f);
  __syncthreads();
  const float d0 = v0 - mu, d1 = v1 - mu, d2 = v2 - mu;
  sd[t] = d0 * d0 + d1 * d1 + d2 * d2;
  __syncthreads();
  for (int o = 128; o > 0; o >>= 1) { if (t < o) sd[t] += sd[t + o]; __syncthreads(); }
  const float rs = rsqrtf(sd[0] * (1.0f / 768.0f) + 1e-5f);

  float dv[3] = {d0, d1, d2};
#pragma unroll
  for (int i = 0; i < 3; ++i) {
    const int e = t + i * 256;
    const float xn = dv[i] * rs * gamma[e];
    if (mode == 0) {
      vt[((size_t)b * E_ + e) * S_ + s] = (_Float16)(xn - avg[e]);
#pragma unroll
      for (int hh = 0; hh < H_; ++hh)
        qh[(((size_t)(b * H_ + hh)) * S_ + s) * E_ + e] = (_Float16)(xn * wqk[hh * E_ + e]);
    } else {
      outp[(size_t)row * E_ + e] = (_Float16)xn;
    }
  }
}

// ---------------- causal softmax (row-wise) ----------------
__global__ __launch_bounds__(256) void softmax_kernel(
    const float* __restrict__ sc, _Float16* __restrict__ attn)
{
  __shared__ float sd[256];
  const int row = blockIdx.x;        // 0..B*H*S-1
  const int s = row & (S_ - 1);
  const int n = s + 1;
  const int t = threadIdx.x;
  const float* x = sc + (size_t)row * S_;
  _Float16* o = attn + (size_t)row * S_;

  float m = -1e30f;
  for (int j = t; j < n; j += 256) m = fmaxf(m, x[j]);
  sd[t] = m;
  __syncthreads();
  for (int off = 128; off > 0; off >>= 1) { if (t < off) sd[t] = fmaxf(sd[t], sd[t + off]); __syncthreads(); }
  m = sd[0];
  __syncthreads();
  float sum = 0.f;
  for (int j = t; j < n; j += 256) sum += expf(x[j] - m);
  sd[t] = sum;
  __syncthreads();
  for (int off = 128; off > 0; off >>= 1) { if (t < off) sd[t] += sd[t + off]; __syncthreads(); }
  const float inv = 1.0f / sd[0];
  for (int j = t; j < S_; j += 256)
    o[j] = (_Float16)(j < n ? expf(x[j] - m) * inv : 0.0f);
}

// ---------------- host orchestration ----------------
extern "C" void kernel_launch(void* const* d_in, const int* in_sizes, int n_in,
                              void* d_out, int out_size, void* d_ws, size_t ws_size,
                              hipStream_t stream)
{
  (void)in_sizes; (void)n_in; (void)out_size; (void)ws_size;
  const int*   xi      = (const int*)  d_in[0];
  const float* wte     = (const float*)d_in[1];
  const float* wpe     = (const float*)d_in[2];
  const float* ln_attn = (const float*)d_in[3];
  const float* w_qk    = (const float*)d_in[4];
  const float* w_o     = (const float*)d_in[5];
  const float* ln_ff   = (const float*)d_in[6];
  const float* w1      = (const float*)d_in[7];
  const float* w2      = (const float*)d_in[8];
  const float* ln_out  = (const float*)d_in[9];
  float* logits = (float*)d_out;

  char* ws = (char*)d_ws;
  size_t off = 0;
  auto alloc = [&](size_t bytes) -> void* {
    void* p = ws + off;
    off = (off + bytes + 255) & ~(size_t)255;
    return p;
  };

  _Float16* wte_h = (_Float16*)alloc((size_t)V_ * E_ * 2);
  _Float16* w_o_h = (_Float16*)alloc((size_t)L_ * E_ * HE_ * 2);
  _Float16* w1_h  = (_Float16*)alloc((size_t)L_ * FF_ * E_ * 2);
  _Float16* w2_h  = (_Float16*)alloc((size_t)L_ * E_ * FF_ * 2);
  float*    avgp  = (float*)   alloc((size_t)E_ * 4);
  float*    hbuf  = (float*)   alloc((size_t)MROWS * E_ * 4);
  _Float16* qh    = (_Float16*)alloc((size_t)B_ * H_ * S_ * E_ * 2);
  _Float16* vt    = (_Float16*)alloc((size_t)B_ * E_ * S_ * 2);
  float*    scor  = (float*)   alloc((size_t)B_ * H_ * S_ * S_ * 4);
  _Float16* attn  = (_Float16*)alloc((size_t)B_ * H_ * S_ * S_ * 2);
  _Float16* ocat  = (_Float16*)alloc((size_t)MROWS * HE_ * 2);
  _Float16* fnh   = (_Float16*)alloc((size_t)MROWS * E_ * 2);
  _Float16* yh    = (_Float16*)alloc((size_t)MROWS * FF_ * 2);
  _Float16* hlnh  = (_Float16*)alloc((size_t)MROWS * E_ * 2);

  // weight conversions f32 -> f16
  f2h_kernel<<<1024, 256, 0, stream>>>(wte, wte_h, (long)V_ * E_);
  f2h_kernel<<<1024, 256, 0, stream>>>(w_o, w_o_h, (long)L_ * E_ * HE_);
  f2h_kernel<<<1024, 256, 0, stream>>>(w1,  w1_h,  (long)L_ * FF_ * E_);
  f2h_kernel<<<1024, 256, 0, stream>>>(w2,  w2_h,  (long)L_ * E_ * FF_);

  avg_kernel<<<(E_ + 255) / 256, 256, 0, stream>>>(wte, avgp);
  embed_kernel<<<(MROWS * E_) / 256, 256, 0, stream>>>(xi, wte, wpe, hbuf);

  const float scl_qk = 0.03608439182435161f;  // 1/sqrt(768)

#define GEMM(MODE, A_, B_, C_, M_, N_, K_, lda_, ldb_, ldc_, scale_)            \
  gemm_nt<MODE><<<dim3((N_) / BNv, (M_) / BMv), 256, 0, stream>>>(              \
      (A_), (B_), (void*)(C_), (M_), (N_), (K_), (lda_), (ldb_), (ldc_), (scale_))

  for (int l = 0; l < L_; ++l) {
    // LN -> per-head Q (f16) and V^T (f16)
    ln_kernel<<<MROWS, 256, 0, stream>>>(hbuf, ln_attn + (size_t)l * E_,
                                         w_qk + (size_t)l * H_ * E_, avgp,
                                         nullptr, qh, vt, 0);
    // scores[b,h] = scale * Q Q^T   (NT, A == B)
    for (int z = 0; z < B_ * H_; ++z)
      GEMM(0, qh + (size_t)z * S_ * E_, qh + (size_t)z * S_ * E_,
           scor + (size_t)z * S_ * S_, S_, S_, E_, E_, E_, S_, scl_qk);
    // causal softmax -> f16 attn
    softmax_kernel<<<B_ * H_ * S_, 256, 0, stream>>>(scor, attn);
    // o[b,h] = (attn @ V) * 1/(s+1), scattered into concat (B,S,H*E)
    for (int z = 0; z < B_ * H_; ++z) {
      int b = z / H_, hh = z % H_;
      GEMM(1, attn + (size_t)z * S_ * S_, vt + (size_t)b * E_ * S_,
           ocat + (size_t)b * S_ * HE_ + (size_t)hh * E_,
           S_, E_, S_, S_, S_, HE_, 1.0f);
    }
    // h += ocat @ W_o^T
    GEMM(2, ocat, w_o_h + (size_t)l * E_ * HE_, hbuf,
         MROWS, E_, HE_, HE_, HE_, E_, 1.0f);
    // FF: LN -> f16
    ln_kernel<<<MROWS, 256, 0, stream>>>(hbuf, ln_ff + (size_t)l * E_,
                                         nullptr, nullptr, fnh, nullptr, nullptr, 1);
    // y = gelu(fn @ W1^T)
    GEMM(3, fnh, w1_h + (size_t)l * FF_ * E_, yh,
         MROWS, FF_, E_, E_, E_, FF_, 1.0f);
    // h += y @ W2^T
    GEMM(2, yh, w2_h + (size_t)l * E_ * FF_, hbuf,
         MROWS, E_, FF_, FF_, FF_, E_, 1.0f);
  }

  // final LN, logits = hln @ wte^T
  ln_kernel<<<MROWS, 256, 0, stream>>>(hbuf, ln_out, nullptr, nullptr,
                                       hlnh, nullptr, nullptr, 1);
  GEMM(0, hlnh, wte_h, logits, MROWS, V_, E_, E_, E_, V_, 1.0f);

#undef GEMM
}
